// CRF_12610023981843
// MI455X (gfx1250) — compile-verified, matrix operand order checked by reading, not run on hardware
//
#include <hip/hip_runtime.h>

typedef __attribute__((ext_vector_type(16))) _Float16 v16h;
typedef __attribute__((ext_vector_type(8)))  float    v8f;
typedef __attribute__((ext_vector_type(4)))  float    v4f;

#define KT 128            // number of tags (K)
#define NEG_VAL -10000.0f
#define LOG2E 1.4426950408889634f
#define LN2   0.6931471805599453f

// raw hardware transcendentals: v_exp_f32 computes 2^x, v_log_f32 computes log2(x)
__device__ __forceinline__ float fast_exp(float x) {
  return __builtin_amdgcn_exp2f(x * LOG2E);
}
__device__ __forceinline__ float fast_log(float x) {
  return __builtin_amdgcn_logf(x) * LN2;
}
// single v_med3_f32 clamp (no canonicalize, unlike fmaxf)
__device__ __forceinline__ float clamp_pos(float x) {
  return __builtin_amdgcn_fmed3f(x, 1e-30f, 3.0e38f);
}

// ---------------------------------------------------------------------------
// Kernel 1: preprocess transition matrix.
//   rowmax[i] = max_j trans[i][j]
//   E16[i][j] = (f16) exp(trans[i][j] - rowmax[i])     (row-major, K x K)
// ---------------------------------------------------------------------------
__global__ void crf_prep(const float* __restrict__ trans,
                         _Float16* __restrict__ E16,
                         float* __restrict__ rowmax) {
  int i = threadIdx.x;                 // one thread per row, 128 threads
  const float* row = trans + i * KT;
  float m = -3.4e38f;
  for (int j = 0; j < KT; ++j) m = fmaxf(m, row[j]);
  rowmax[i] = m;
  for (int j = 0; j < KT; ++j)
    E16[i * KT + j] = (_Float16)fast_exp(row[j] - m);
}

// ---------------------------------------------------------------------------
// Kernel 2: CRF forward scan. One workgroup = 16 batch elements, 256 threads
// (8 wave32). Wave w owns output-tag rows [16w, 16w+16). A-matrix (exp of
// transitions) lives in VGPRs for the whole scan; B fragments (exp of scores)
// are rebuilt in LDS each step in WMMA fragment layout.
// ---------------------------------------------------------------------------
__launch_bounds__(256, 1)
__global__ void crf_main(const float* __restrict__ h,        // [B,T,K]
                         const float* __restrict__ trans,    // [K,K]
                         const int*   __restrict__ lengths,  // [B]
                         const _Float16* __restrict__ E16,   // [K,K] f16
                         const float* __restrict__ rowmaxG,  // [K]
                         float* __restrict__ out,            // [B]
                         int T) {
  __shared__ float scoreS[KT * 17];                     // [j][b], stride 17 (bank spread)
  __shared__ __align__(16) float pmaxS[16 * 16];        // [b][chunk] partial maxes
  __shared__ __align__(32) _Float16 fragB[4 * 32 * 16]; // 4 K-frags x 32 lanes x 16 halves
  __shared__ float tEndS[KT];                           // trans[END][:]

  const int tid   = threadIdx.x;
  const int w     = tid >> 5;        // wave id 0..7  -> M tile
  const int l     = tid & 31;        // lane
  const int g     = l >> 4;          // lane half
  const int b     = tid & 15;        // batch lane (same in all stages)
  const int chunk = tid >> 4;        // 0..15 -> j block of 8
  const int j0    = chunk * 8;
  const int bBase = blockIdx.x * 16;
  const int iBase = 16 * w + 8 * g;  // first of this thread's 8 C rows

  // ---- load resident A fragments: E16 rows [16w, 16w+16), all K ----
  v16h aF[4];
  {
    int M = l & 15;
    for (int kb = 0; kb < 4; ++kb) {
      union { v16h v; _Float16 f[16]; } u;
      for (int vv = 0; vv < 8; ++vv) {
        // 16-bit A layout: VGPR vv holds K pair; K = (vv/4)*16 + (vv%4)*2 + p + 8*g
        int k = (vv >> 2) * 16 + (vv & 3) * 2 + 8 * g + kb * 32;
        const _Float16* p = E16 + (16 * w + M) * KT + k;
        u.f[2 * vv]     = p[0];
        u.f[2 * vv + 1] = p[1];
      }
      aF[kb] = u.v;
    }
  }

  // per-thread row maxima for its 8 C rows
  float rmx[8];
  for (int r = 0; r < 8; ++r) rmx[r] = rowmaxG[iBase + r];

  const int lenb = lengths[bBase + b];

  if (tid < KT) tEndS[tid] = trans[(KT - 2) * KT + tid];   // END row

  // init: score0 = NEG everywhere, 0 at START = K-1
  for (int idx = tid; idx < KT * 16; idx += 256) {
    int j = idx >> 4, bb = idx & 15;
    scoreS[j * 17 + bb] = (j == KT - 1) ? 0.0f : NEG_VAL;
  }
  __syncthreads();

  // h element pointer for this thread's (b, iBase..iBase+7) C slots
  const float* hPtr = h + (size_t)(bBase + b) * (size_t)T * KT + iBase;

  for (int t = 0; t < T; ++t) {
    // prefetch h_t early (8 contiguous floats -> 2x b128 global loads)
    v4f h0 = *(const v4f*)(hPtr + (size_t)t * KT);
    v4f h1 = *(const v4f*)(hPtr + (size_t)t * KT + 4);

    // ---- stage A: per-batch partial max over 8 j's ----
    float vals[8];
    for (int r = 0; r < 8; ++r) vals[r] = scoreS[(j0 + r) * 17 + b];
    float pm = vals[0];
    for (int r = 1; r < 8; ++r) pm = fmaxf(pm, vals[r]);
    pmaxS[b * 16 + chunk] = pm;
    __syncthreads();

    // ---- stage B: full max m[b] (4x ds_load_b128), exp, write B fragment ----
    float m;
    {
      const v4f* pv = (const v4f*)(pmaxS + b * 16);
      v4f p0 = pv[0], p1 = pv[1], p2 = pv[2], p3 = pv[3];
      v4f q0 = {fmaxf(p0.x, p1.x), fmaxf(p0.y, p1.y),
                fmaxf(p0.z, p1.z), fmaxf(p0.w, p1.w)};
      v4f q1 = {fmaxf(p2.x, p3.x), fmaxf(p2.y, p3.y),
                fmaxf(p2.z, p3.z), fmaxf(p2.w, p3.w)};
      m = fmaxf(fmaxf(fmaxf(q0.x, q1.x), fmaxf(q0.y, q1.y)),
                fmaxf(fmaxf(q0.z, q1.z), fmaxf(q0.w, q1.w)));
    }
    {
      union { _Float16 f[8]; v4f v; } e;
      for (int r = 0; r < 8; ++r) e.f[r] = (_Float16)fast_exp(vals[r] - m);
      // j0 = chunk*8 maps to: frag kb, lane, half block hi -> contiguous 8 halves
      int kb   = chunk >> 2;
      int lane = ((chunk & 1) << 4) + b;
      int hi   = (chunk >> 1) & 1;
      *(v4f*)(fragB + kb * 512 + lane * 16 + hi * 8) = e.v;  // one ds_store_b128
    }
    __syncthreads();

    // ---- stage C: issue ALL 8 ds_load_b128 first, then 4 back-to-back
    //      WMMAs. Tied "+v" asm operands force all 4 fragments (32 VGPRs)
    //      live at once -> loads pipeline behind a single dscnt wait.
    v16h bF0 = *(const v16h*)(fragB + 0 * 512 + l * 16);
    v16h bF1 = *(const v16h*)(fragB + 1 * 512 + l * 16);
    v16h bF2 = *(const v16h*)(fragB + 2 * 512 + l * 16);
    v16h bF3 = *(const v16h*)(fragB + 3 * 512 + l * 16);
    asm volatile("" : "+v"(bF0), "+v"(bF1), "+v"(bF2), "+v"(bF3));
    v8f c8 = {0.f, 0.f, 0.f, 0.f, 0.f, 0.f, 0.f, 0.f};
    c8 = __builtin_amdgcn_wmma_f32_16x16x32_f16(false, aF[0], false, bF0,
                                                (short)0, c8, false, false);
    c8 = __builtin_amdgcn_wmma_f32_16x16x32_f16(false, aF[1], false, bF1,
                                                (short)0, c8, false, false);
    c8 = __builtin_amdgcn_wmma_f32_16x16x32_f16(false, aF[2], false, bF2,
                                                (short)0, c8, false, false);
    c8 = __builtin_amdgcn_wmma_f32_16x16x32_f16(false, aF[3], false, bF3,
                                                (short)0, c8, false, false);

    float hv[8] = {h0.x, h0.y, h0.z, h0.w, h1.x, h1.y, h1.z, h1.w};
    float snew[8];
    for (int r = 0; r < 8; ++r)
      snew[r] = fast_log(clamp_pos(c8[r])) + rmx[r] + m + hv[r];
    if (t >= lenb) {     // masked step: freeze this batch's scores
      for (int r = 0; r < 8; ++r) snew[r] = scoreS[(iBase + r) * 17 + b];
    }
    // each thread reads/writes only its own (i,b) cells -> no extra barrier
    for (int r = 0; r < 8; ++r) scoreS[(iBase + r) * 17 + b] = snew[r];
    __syncthreads();
  }

  // ---- final: out[b] = logsumexp_j(score[j][b] + trans[END][j]) ----
  if (tid < 16) {
    float mm = -3.4e38f;
    for (int j = 0; j < KT; ++j)
      mm = fmaxf(mm, scoreS[j * 17 + tid] + tEndS[j]);
    float s = 0.f;
    for (int j = 0; j < KT; ++j)
      s += fast_exp(scoreS[j * 17 + tid] + tEndS[j] - mm);
    out[bBase + tid] = fast_log(s) + mm;
  }
}

// ---------------------------------------------------------------------------
extern "C" void kernel_launch(void* const* d_in, const int* in_sizes, int n_in,
                              void* d_out, int out_size, void* d_ws, size_t ws_size,
                              hipStream_t stream) {
  const float* h      = (const float*)d_in[0];   // [B,T,K] f32
  const float* trans  = (const float*)d_in[1];   // [K,K]   f32
  const int*   length = (const int*)d_in[2];     // [B]     i32

  float* out = (float*)d_out;                    // [B] f32

  const int B = in_sizes[2];
  const int T = in_sizes[0] / (B * KT);

  _Float16* E16   = (_Float16*)d_ws;                                    // 32 KB
  float*    rowmx = (float*)((char*)d_ws + KT * KT * sizeof(_Float16)); // 512 B

  crf_prep<<<1, KT, 0, stream>>>(trans, E16, rowmx);
  crf_main<<<B / 16, 256, 0, stream>>>(h, trans, length, E16, rowmx, out, T);
}
